// SpConvNet_9758165696718
// MI455X (gfx1250) — matrix operand order repaired and checked by reading, not compiled, not run on hardware
//
#include <hip/hip_runtime.h>
#include <hip/hip_bf16.h>

typedef __attribute__((ext_vector_type(16))) _Float16 v16h;
typedef __attribute__((ext_vector_type(8)))  float    v8f;

#define BATCH 2
#define CIN   32
#define COUT  64
#define D0    41
#define O1    39
#define OT    79
#define ODN   77
#define EPSV  1e-5f

// ---------------------------------------------------------------- utilities
__global__ void zero_words_k(unsigned int* p, size_t n) {
    size_t i = (size_t)blockIdx.x * blockDim.x + threadIdx.x;
    for (; i < n; i += (size_t)gridDim.x * blockDim.x) p[i] = 0u;
}

__global__ void init_counts_k(float* cnt) {
    if (threadIdx.x == 0) {
        cnt[0] = 0.0f;                                   // n1 (39^3 mask count)
        cnt[1] = 0.0f;                                   // nt (79^3 mask count)
        cnt[2] = (float)(BATCH * ODN * ODN * ODN);       // dense count
    }
}

__global__ void zero_stats_k(float* stats) {
    if (threadIdx.x < 128) stats[threadIdx.x] = 0.0f;    // sum[64], sumsq[64]
}

// ---------------------------------------------------------------- scatter
__global__ void scatter_k(const float* __restrict__ feat,
                          const int* __restrict__ coors,
                          _Float16* __restrict__ x0, float* __restrict__ m0,
                          int npts) {
    int i = blockIdx.x * blockDim.x + threadIdx.x;
    if (i >= npts) return;
    int b = coors[4 * i + 0], z = coors[4 * i + 1];
    int y = coors[4 * i + 2], x = coors[4 * i + 3];
    size_t cell = (((size_t)b * D0 + z) * D0 + y) * D0 + x;
    m0[cell] = 1.0f;
    #pragma unroll 8
    for (int c = 0; c < CIN; ++c)
        x0[cell * CIN + c] = (_Float16)feat[(size_t)i * CIN + c];
}

// -------------------------------------------------- masks (shrink / dilate)
__global__ void mask_shrink_k(const float* __restrict__ m0,
                              float* __restrict__ m1, float* __restrict__ cnt) {
    const int npos = BATCH * O1 * O1 * O1;
    int i = blockIdx.x * blockDim.x + threadIdx.x;
    float val = 0.0f;
    if (i < npos) {
        int r = i;
        int b = r / (O1 * O1 * O1); r %= O1 * O1 * O1;
        int oz = r / (O1 * O1);     r %= O1 * O1;
        int oy = r / O1;            int ox = r % O1;
        bool any = false;
        for (int tz = 0; tz < 3; ++tz)
            for (int ty = 0; ty < 3; ++ty)
                for (int tx = 0; tx < 3; ++tx) {
                    size_t a = (((size_t)b * D0 + oz + tz) * D0 + oy + ty) * D0 + ox + tx;
                    any = any || (m0[a] > 0.0f);
                }
        val = any ? 1.0f : 0.0f;
        m1[i] = val;
    }
    __shared__ float s[256];
    s[threadIdx.x] = val;
    __syncthreads();
    for (int st = 128; st > 0; st >>= 1) {
        if (threadIdx.x < st) s[threadIdx.x] += s[threadIdx.x + st];
        __syncthreads();
    }
    if (threadIdx.x == 0 && s[0] != 0.0f) atomicAdd(cnt, s[0]);
}

__global__ void mask_dilate_k(const float* __restrict__ m1,
                              float* __restrict__ mt, float* __restrict__ cnt) {
    const int npos = BATCH * OT * OT * OT;
    int i = blockIdx.x * blockDim.x + threadIdx.x;
    float val = 0.0f;
    if (i < npos) {
        int r = i;
        int b = r / (OT * OT * OT); r %= OT * OT * OT;
        int oz = r / (OT * OT);     r %= OT * OT;
        int oy = r / OT;            int ox = r % OT;
        bool any = false;
        for (int tz = 0; tz < 3; ++tz) {
            int uz = oz + tz - 2;
            if (uz < 0 || (uz & 1) || (uz >> 1) >= O1) continue;
            for (int ty = 0; ty < 3; ++ty) {
                int uy = oy + ty - 2;
                if (uy < 0 || (uy & 1) || (uy >> 1) >= O1) continue;
                for (int tx = 0; tx < 3; ++tx) {
                    int ux = ox + tx - 2;
                    if (ux < 0 || (ux & 1) || (ux >> 1) >= O1) continue;
                    size_t a = (((size_t)b * O1 + (uz >> 1)) * O1 + (uy >> 1)) * O1 + (ux >> 1);
                    any = any || (m1[a] > 0.0f);
                }
            }
        }
        val = any ? 1.0f : 0.0f;
        mt[i] = val;
    }
    __shared__ float s[256];
    s[threadIdx.x] = val;
    __syncthreads();
    for (int st = 128; st > 0; st >>= 1) {
        if (threadIdx.x < st) s[threadIdx.x] += s[threadIdx.x + st];
        __syncthreads();
    }
    if (threadIdx.x == 0 && s[0] != 0.0f) atomicAdd(cnt, s[0]);
}

// ------------------------------------------- weight swizzle into A-fragment
// A-matrix (16x32 f16) lane layout (ISA 7.12.2): lane m = lane&15, hi = lane>>4
// this code's order (B matches): element e maps to k = (hi<<4) + e
// memory k-index is tap-major: k = t*CK + ci  (t in 0..26 over 3x3x3)
// Packed kk-major: wp[((kk*4 + co_tile)*32 + lane)*16 + e]
__global__ void prep_weights_k(const float* __restrict__ w, _Float16* __restrict__ wp,
                               int CK, int nch) {
    int i = blockIdx.x * blockDim.x + threadIdx.x;
    int total = nch * 4 * 32 * 16;
    if (i >= total) return;
    int e    = i & 15;
    int lane = (i >> 4) & 31;
    int cot  = (i >> 9) & 3;
    int kk   = i >> 11;
    int m = lane & 15, hi = lane >> 4;
    int j = e >> 1, s = e & 1;
    int kloc = (hi << 4) + e;
    int k  = kk * 32 + kloc;
    int t  = k / CK, ci = k % CK;
    int co = cot * 16 + m;
    wp[i] = (_Float16)w[((size_t)co * CK + ci) * 27 + t];
}

// ------------------------------------------------ implicit-GEMM WMMA conv
// One wave computes a 64(co) x 32(pos) tile: 4 co-tiles x 2 n-tiles.
// Per K-chunk: 2 B fragments + 4 A fragments feed 8 wmmas (A reused 2x,
// B reused 4x) -> 0.75 KB of cache traffic per 16-KFLOP wmma.
template <int ID, int OD, int CK, int PAD, int TRANS>
__global__ __launch_bounds__(256) void conv_wmma_k(
        const _Float16* __restrict__ xin,
        const _Float16* __restrict__ wp,
        const float* __restrict__ bias,
        const float* __restrict__ mask,
        float* __restrict__ out) {
    constexpr int NCH  = (CK * 27) / 32;
    constexpr int ODHW = OD * OD * OD;
    const int Ntot  = BATCH * ODHW;
    const int tiles = (Ntot + 31) >> 5;
    const int wave  = (int)((blockIdx.x * blockDim.x + threadIdx.x) >> 5);
    if (wave >= tiles) return;                          // wave-uniform exit
    const int lane = threadIdx.x & 31;
    const int hi = lane >> 4, lm = lane & 15;

    int nL[2], bb[2], oz[2], oy[2], ox[2];
    #pragma unroll
    for (int p = 0; p < 2; ++p) {
        nL[p] = wave * 32 + p * 16 + lm;
        int r = nL[p] < Ntot ? nL[p] : Ntot - 1;
        bb[p] = r / ODHW;      r %= ODHW;               // compile-time divisors
        oz[p] = r / (OD * OD); r %= OD * OD;
        oy[p] = r / OD;
        ox[p] = r % OD;
    }

    v8f acc[4][2] = {{{}, {}}, {{}, {}}, {{}, {}}, {{}, {}}};
    #pragma unroll
    for (int kk = 0; kk < NCH; ++kk) {
        const int t   = (kk * 32) / CK;                 // whole chunk = one tap
        const int ci0 = (kk * 32) % CK + hi * 16;
        const int tz = t / 9, tq = t % 9, ty = tq / 3, tx = tq % 3;
        v16h bfrag[2];
        #pragma unroll
        for (int p = 0; p < 2; ++p) {
            bool ok; int iz, iy, ix;
            if (TRANS) {
                const int uz = oz[p] + tz - 2, uy = oy[p] + ty - 2, ux = ox[p] + tx - 2;
                iz = uz >> 1; iy = uy >> 1; ix = ux >> 1;
                ok = uz >= 0 && uy >= 0 && ux >= 0 &&
                     !(uz & 1) && !(uy & 1) && !(ux & 1) &&
                     iz < ID && iy < ID && ix < ID;
            } else {
                iz = oz[p] + tz - PAD; iy = oy[p] + ty - PAD; ix = ox[p] + tx - PAD;
                ok = iz >= 0 && iy >= 0 && ix >= 0 && iz < ID && iy < ID && ix < ID;
            }
            v16h bf = {};
            if (ok) {
                const _Float16* src =
                    xin + ((((size_t)bb[p] * ID + iz) * ID + iy) * ID + ix) * CK + ci0;
                bf = *(const v16h*)src;
            }
            bfrag[p] = bf;
        }
        #pragma unroll
        for (int cot = 0; cot < 4; ++cot) {
            const v16h afrag =
                *(const v16h*)(wp + (((size_t)kk * 4 + cot) * 32 + lane) * 16);
            #pragma unroll
            for (int p = 0; p < 2; ++p)
                acc[cot][p] = __builtin_amdgcn_wmma_f32_16x16x32_f16(
                                  false, afrag, false, bfrag[p], (short)0,
                                  acc[cot][p], false, false);
        }
    }
    #pragma unroll
    for (int p = 0; p < 2; ++p) {
        if (nL[p] < Ntot) {
            const float mv = mask ? mask[nL[p]] : 1.0f;
            #pragma unroll
            for (int cot = 0; cot < 4; ++cot) {
                v8f o;
                #pragma unroll
                for (int rr = 0; rr < 8; ++rr)
                    o[rr] = (acc[cot][p][rr] + bias[cot * 16 + hi * 8 + rr]) * mv;
                *(v8f*)(out + (size_t)nL[p] * COUT + cot * 16 + hi * 8) = o;
            }
        }
    }
}

// ------------------------------------------------------------ BN reductions
__global__ void bn_stats_k(const float* __restrict__ act, float* __restrict__ stats,
                           int npos) {
    __shared__ float ss[256], sq[256];
    const int c = threadIdx.x & 63;
    const int g = threadIdx.x >> 6;
    float a = 0.0f, q = 0.0f;
    for (int p = blockIdx.x * 4 + g; p < npos; p += gridDim.x * 4) {
        const float v = act[(size_t)p * COUT + c];
        a += v; q += v * v;
    }
    ss[threadIdx.x] = a; sq[threadIdx.x] = q;
    __syncthreads();
    if (threadIdx.x < 64) {
        a = ss[c] + ss[c + 64] + ss[c + 128] + ss[c + 192];
        q = sq[c] + sq[c + 64] + sq[c + 128] + sq[c + 192];
        atomicAdd(&stats[c], a);
        atomicAdd(&stats[64 + c], q);
    }
}

__global__ void bn_norm_k(const float* __restrict__ act,
                          const float* __restrict__ mask,
                          const float* __restrict__ stats,
                          const float* __restrict__ nptr,
                          const float* __restrict__ gamma,
                          const float* __restrict__ beta,
                          _Float16* __restrict__ out16,
                          float* __restrict__ out32,
                          int npos) {
    const size_t total = (size_t)npos * COUT;
    size_t i = (size_t)blockIdx.x * blockDim.x + threadIdx.x;
    const float n = *nptr;
    for (; i < total; i += (size_t)gridDim.x * blockDim.x) {
        const int c = (int)(i & 63);
        const float mu  = stats[c] / n;
        const float var = stats[64 + c] / n - mu * mu;
        const float inv = gamma[c] * rsqrtf(var + EPSV);
        const float m   = mask ? mask[i >> 6] : 1.0f;
        float y = (act[i] - mu * m) * inv + beta[c] * m;
        y = fmaxf(y, 0.0f);
        if (out16) out16[i] = (_Float16)y;
        if (out32) out32[i] = y;
    }
}

// ---------------------------------------------------------------- launcher
static inline size_t alignup(size_t v) { return (v + 255) & ~(size_t)255; }

extern "C" void kernel_launch(void* const* d_in, const int* in_sizes, int n_in,
                              void* d_out, int out_size, void* d_ws, size_t ws_size,
                              hipStream_t stream) {
    const float* feats = (const float*)d_in[0];
    const int*   coors = (const int*)d_in[1];
    const float* w1 = (const float*)d_in[3];  const float* b1 = (const float*)d_in[4];
    const float* w2 = (const float*)d_in[5];  const float* b2 = (const float*)d_in[6];
    const float* w3 = (const float*)d_in[7];  const float* b3 = (const float*)d_in[8];
    const float* wt = (const float*)d_in[9];  const float* bt = (const float*)d_in[10];
    const float* wd = (const float*)d_in[11]; const float* bd = (const float*)d_in[12];
    const float* g1 = (const float*)d_in[13]; const float* be1 = (const float*)d_in[14];
    const float* g2 = (const float*)d_in[15]; const float* be2 = (const float*)d_in[16];
    const float* g3 = (const float*)d_in[17]; const float* be3 = (const float*)d_in[18];
    const float* gt = (const float*)d_in[19]; const float* bet = (const float*)d_in[20];
    const float* gd = (const float*)d_in[21]; const float* bed = (const float*)d_in[22];
    const int npts = in_sizes[0] / CIN;

    const int N0 = BATCH * D0 * D0 * D0;      // 41^3 cells
    const int N1 = BATCH * O1 * O1 * O1;      // 39^3 positions
    const int NT = BATCH * OT * OT * OT;      // 79^3 positions
    const int ND = BATCH * ODN * ODN * ODN;   // 77^3 positions

    // workspace carve-up
    char* ws = (char*)d_ws; size_t off = 0;
    _Float16* x0h = (_Float16*)(ws + off); off = alignup(off + (size_t)N0 * CIN * 2);
    float*    m0  = (float*)(ws + off);    off = alignup(off + (size_t)N0 * 4);
    float*    m1  = (float*)(ws + off);    off = alignup(off + (size_t)N1 * 4);
    float*    mt  = (float*)(ws + off);    off = alignup(off + (size_t)NT * 4);
    float*    cnt = (float*)(ws + off);    off = alignup(off + 16 * 4);
    float*    st  = (float*)(ws + off);    off = alignup(off + 128 * 4);
    _Float16* wp1 = (_Float16*)(ws + off); off = alignup(off + (size_t)COUT * CIN * 27 * 2);
    _Float16* wp2 = (_Float16*)(ws + off); off = alignup(off + (size_t)COUT * COUT * 27 * 2);
    _Float16* wp3 = (_Float16*)(ws + off); off = alignup(off + (size_t)COUT * COUT * 27 * 2);
    _Float16* wpt = (_Float16*)(ws + off); off = alignup(off + (size_t)COUT * COUT * 27 * 2);
    _Float16* wpd = (_Float16*)(ws + off); off = alignup(off + (size_t)COUT * COUT * 27 * 2);
    float*    actA = (float*)(ws + off);   off = alignup(off + (size_t)NT * COUT * 4);
    _Float16* actH = (_Float16*)(ws + off); off = alignup(off + (size_t)NT * COUT * 2);

    // 0) zero scatter targets, init counts
    {
        size_t w0 = (size_t)N0 * CIN * 2 / 4;
        zero_words_k<<<1024, 256, 0, stream>>>((unsigned int*)x0h, w0);
        zero_words_k<<<256, 256, 0, stream>>>((unsigned int*)m0, (size_t)N0);
        init_counts_k<<<1, 32, 0, stream>>>(cnt);
    }
    // 1) scatter features + mask
    scatter_k<<<(npts + 255) / 256, 256, 0, stream>>>(feats, coors, x0h, m0, npts);
    // 2) swizzle weights into WMMA A-fragment layout
    prep_weights_k<<<(27 * 4 * 512 + 255) / 256, 256, 0, stream>>>(w1, wp1, CIN, 27);
    prep_weights_k<<<(54 * 4 * 512 + 255) / 256, 256, 0, stream>>>(w2, wp2, COUT, 54);
    prep_weights_k<<<(54 * 4 * 512 + 255) / 256, 256, 0, stream>>>(w3, wp3, COUT, 54);
    prep_weights_k<<<(54 * 4 * 512 + 255) / 256, 256, 0, stream>>>(wt, wpt, COUT, 54);
    prep_weights_k<<<(54 * 4 * 512 + 255) / 256, 256, 0, stream>>>(wd, wpd, COUT, 54);
    // 3) masks
    mask_shrink_k<<<(N1 + 255) / 256, 256, 0, stream>>>(m0, m1, &cnt[0]);
    mask_dilate_k<<<(NT + 255) / 256, 256, 0, stream>>>(m1, mt, &cnt[1]);

    auto gemm_blocks = [](int ntot) { int tiles = (ntot + 31) / 32; return (tiles + 7) / 8; };

    // stage 1: conv1 (41^3 -> 39^3), BN(m1), ReLU
    conv_wmma_k<D0, O1, CIN, 0, 0><<<gemm_blocks(N1), 256, 0, stream>>>(
        x0h, wp1, b1, m1, actA);
    zero_stats_k<<<1, 128, 0, stream>>>(st);
    bn_stats_k<<<1024, 256, 0, stream>>>(actA, st, N1);
    bn_norm_k<<<4096, 256, 0, stream>>>(actA, m1, st, &cnt[0], g1, be1, actH, nullptr, N1);

    // stage 2: conv2 (39^3 pad1), BN(m1), ReLU
    conv_wmma_k<O1, O1, COUT, 1, 0><<<gemm_blocks(N1), 256, 0, stream>>>(
        actH, wp2, b2, m1, actA);
    zero_stats_k<<<1, 128, 0, stream>>>(st);
    bn_stats_k<<<1024, 256, 0, stream>>>(actA, st, N1);
    bn_norm_k<<<4096, 256, 0, stream>>>(actA, m1, st, &cnt[0], g2, be2, actH, nullptr, N1);

    // stage 3: conv3 (39^3 pad1), BN(m1), ReLU
    conv_wmma_k<O1, O1, COUT, 1, 0><<<gemm_blocks(N1), 256, 0, stream>>>(
        actH, wp3, b3, m1, actA);
    zero_stats_k<<<1, 128, 0, stream>>>(st);
    bn_stats_k<<<1024, 256, 0, stream>>>(actA, st, N1);
    bn_norm_k<<<4096, 256, 0, stream>>>(actA, m1, st, &cnt[0], g3, be3, actH, nullptr, N1);

    // stage 4: transposed conv (39^3 -> 79^3), BN(mt), ReLU
    conv_wmma_k<O1, OT, COUT, 2, 1><<<gemm_blocks(NT), 256, 0, stream>>>(
        actH, wpt, bt, mt, actA);
    zero_stats_k<<<1, 128, 0, stream>>>(st);
    bn_stats_k<<<1024, 256, 0, stream>>>(actA, st, NT);
    bn_norm_k<<<8192, 256, 0, stream>>>(actA, mt, st, &cnt[1], gt, bet, actH, nullptr, NT);

    // stage 5: conv d (79^3 -> 77^3, no mask), dense BN, ReLU -> d_out (f32)
    conv_wmma_k<OT, ODN, COUT, 0, 0><<<gemm_blocks(ND), 256, 0, stream>>>(
        actH, wpd, bd, nullptr, actA);
    zero_stats_k<<<1, 128, 0, stream>>>(st);
    bn_stats_k<<<1024, 256, 0, stream>>>(actA, st, ND);
    bn_norm_k<<<8192, 256, 0, stream>>>(actA, nullptr, st, &cnt[2], gd, bed,
                                        nullptr, (float*)d_out, ND);
    (void)n_in; (void)out_size; (void)ws_size; (void)in_sizes;
}
